// LieSelfAttention_56315611185335
// MI455X (gfx1250) — matrix operand order repaired and checked by reading, not compiled, not run on hardware
//
#include <hip/hip_runtime.h>
#include <hip/hip_bf16.h>
#include <math.h>

// ---------------------------------------------------------------------------
// Problem constants (from the reference)
// ---------------------------------------------------------------------------
#define BB      4
#define NN      2048
#define DD      6
#define CC      128
#define HEADS   8
#define DHEAD   64
#define INNER   512          // HEADS * DHEAD
#define KS      32           // MC_SAMPLES
#define RADIUS  2.0f

typedef __attribute__((ext_vector_type(16))) _Float16 v16h;
typedef __attribute__((ext_vector_type(8)))  _Float16 v8h;
typedef __attribute__((ext_vector_type(2)))  _Float16 v2h;
typedef __attribute__((ext_vector_type(8)))  float    v8f;

// ---------------------------------------------------------------------------
// WMMA fragment helpers for V_WMMA_F32_16X16X32_F16 (wave32).
//
// A (16x32 f16, row-major MxK): lane L holds row M = L%16.  Its 16 halfs are
//   K = k0 + (L/16)*16 + {0..15}
//   -> one contiguous 32-byte run -> two b128 loads.
// B (32x16 f16): lane L holds column N = L%16, K = (L/16)*16 + {0..15}.
//   With B stored TRANSPOSED (BT, N x K row-major) this is one contiguous
//   32-byte run -> two b128 loads.
// C/D (16x16 f32): lane L holds N = L%16, VGPR v holds M = v + (L/16)*8.
// ---------------------------------------------------------------------------
__device__ __forceinline__ v16h load_a_frag(const _Float16* A, int lda,
                                            int m0, int k0, int lane) {
  const int m  = m0 + (lane & 15);
  const int kh = (lane >> 4) * 16;
  const _Float16* p = A + (size_t)m * lda + k0 + kh;
  const v8h lo = *(const v8h*)(p);
  const v8h hi = *(const v8h*)(p + 8);
  return __builtin_shufflevector(lo, hi, 0, 1, 2, 3, 4, 5, 6, 7,
                                         8, 9, 10, 11, 12, 13, 14, 15);
}

__device__ __forceinline__ v16h load_bT_frag(const _Float16* BT, int ldk,
                                             int k0, int n0, int lane) {
  const int n  = n0 + (lane & 15);
  const int kg = (lane >> 4) * 16;
  return *(const v16h*)(BT + (size_t)n * ldk + k0 + kg);
}

__device__ __forceinline__ v8f wmma_f16(v16h a, v16h b, v8f c) {
  return __builtin_amdgcn_wmma_f32_16x16x32_f16(false, a, false, b,
                                                (short)0, c, false, false);
}

__device__ __forceinline__ void store_acc_f32(float* C, int ldc, int m0,
                                              int n0, int lane, v8f c) {
  const int n  = n0 + (lane & 15);
  const int mb = m0 + (lane >> 4) * 8;
#pragma unroll
  for (int v = 0; v < 8; ++v) C[(size_t)(mb + v) * ldc + n] = c[v];
}

// ---------------------------------------------------------------------------
// Kernel 0: f32 -> f16 conversions; weights stored TRANSPOSED (N x K);
// mask output tail.   grid: 4096 x 256
// ---------------------------------------------------------------------------
__global__ void prep_kernel(const float* __restrict__ vals,
                            const float* __restrict__ Wq,
                            const float* __restrict__ Wk,
                            const float* __restrict__ Wv,
                            const float* __restrict__ Wo,
                            const unsigned char* __restrict__ mask,
                            _Float16* __restrict__ vals_h,
                            _Float16* __restrict__ WqT,
                            _Float16* __restrict__ WkT,
                            _Float16* __restrict__ WvT,
                            _Float16* __restrict__ WoT,
                            float* __restrict__ mask_out) {
  const int i = blockIdx.x * blockDim.x + threadIdx.x;
  if (i < BB * NN * CC) vals_h[i] = (_Float16)vals[i];
  if (i < CC * INNER) {
    // Wq/Wk/Wv are (CC x INNER): element i = [r][c], r = i/INNER, c = i%INNER
    const int r = i >> 9;            // / INNER
    const int c = i & (INNER - 1);
    WqT[(size_t)c * CC + r] = (_Float16)Wq[i];
    WkT[(size_t)c * CC + r] = (_Float16)Wk[i];
    WvT[(size_t)c * CC + r] = (_Float16)Wv[i];
    // Wo is (INNER x CC): element i = [r2][c2], r2 = i/CC, c2 = i%CC
    const int r2 = i >> 7;           // / CC
    const int c2 = i & (CC - 1);
    WoT[(size_t)c2 * INNER + r2] = (_Float16)Wo[i];
  }
  if (i < BB * NN) mask_out[i] = mask[i] ? 1.0f : 0.0f;
}

// ---------------------------------------------------------------------------
// Kernel 1: distances + randomized ball-query top-32, fused with the
// pass-through copy of pairs_abq (read once, write once, NT hints).
// One block (256 threads) per (b, i) query row.
// ---------------------------------------------------------------------------
__global__ void topk_kernel(const float* __restrict__ pairs,
                            const unsigned char* __restrict__ mask,
                            const float* __restrict__ noise,
                            float* __restrict__ pairs_out,
                            int*   __restrict__ nbhd_idx,
                            float* __restrict__ nbhd_maskf) {
  __shared__ float sc[NN];
  __shared__ float rv[256];
  __shared__ int   ri[256];

  const int qi  = blockIdx.x;        // b*NN + i
  const int b   = qi >> 11;
  const int tid = threadIdx.x;
  const bool mi = mask[qi] != 0;

  const float* prow = pairs + (size_t)qi * NN * DD;
  float*       orow = pairs_out + (size_t)qi * NN * DD;
  const float* nrow = noise + (size_t)qi * NN;

  // scores + streaming copy of the pairs row
  for (int j = tid; j < NN; j += 256) {
    const float* p = prow + (size_t)j * DD;
    float ss = 0.0f;
#pragma unroll
    for (int d = 0; d < DD; ++d) {
      const float x = __builtin_nontemporal_load(p + d);
      __builtin_nontemporal_store(x, orow + (size_t)j * DD + d);
      ss += x * x;
    }
    float dist = sqrtf(ss);
    const bool mj = mask[(b << 11) + j] != 0;
    if (!mj) dist = 1e8f;
    const float wb = (dist < RADIUS && mj && mi) ? 1.0f : 0.0f;
    sc[j] = wb + nrow[j];
  }
  __syncthreads();

  // 32 rounds of block-wide argmax with extraction
  for (int r = 0; r < KS; ++r) {
    float best = -1e30f;
    int   bidx = 0;
    for (int j = tid; j < NN; j += 256) {
      const float v = sc[j];
      if (v > best) { best = v; bidx = j; }
    }
    rv[tid] = best;
    ri[tid] = bidx;
    __syncthreads();
    for (int s = 128; s > 0; s >>= 1) {
      if (tid < s) {
        const bool take = (rv[tid + s] > rv[tid]) ||
                          (rv[tid + s] == rv[tid] && ri[tid + s] < ri[tid]);
        if (take) { rv[tid] = rv[tid + s]; ri[tid] = ri[tid + s]; }
      }
      __syncthreads();
    }
    if (tid == 0) {
      const int   m = ri[0];
      const float v = rv[0];
      nbhd_idx[(size_t)qi * KS + r] = m;
      const bool mm = mask[(b << 11) + m] != 0;
      nbhd_maskf[(size_t)qi * KS + r] = (v > 1.0f && mm) ? 1.0f : 0.0f;
      sc[m] = -1e30f;
    }
    __syncthreads();
  }
}

// ---------------------------------------------------------------------------
// Kernel 2: f16 GEMM  C(f32, MxN) = A(MxK) * BT(NxK)^T
// Each wave computes a 16x64 tile: 1 A fragment reused by 4 B fragments.
// ---------------------------------------------------------------------------
__global__ void gemm_f16_wmma(const _Float16* __restrict__ A,
                              const _Float16* __restrict__ BT,
                              float* __restrict__ C,
                              int M, int Kd, int N) {
  const int lane    = threadIdx.x & 31;
  const int wave    = threadIdx.x >> 5;
  const int tiles_n = N >> 6;                       // 64-wide tiles
  const int tile    = blockIdx.x * (blockDim.x >> 5) + wave;
  if (tile >= (M >> 4) * tiles_n) return;           // uniform per-wave
  const int m0 = (tile / tiles_n) << 4;
  const int n0 = (tile % tiles_n) << 6;

  v8f acc0 = {}, acc1 = {}, acc2 = {}, acc3 = {};
  for (int k0 = 0; k0 < Kd; k0 += 32) {
    const v16h a  = load_a_frag(A, Kd, m0, k0, lane);
    const v16h b0 = load_bT_frag(BT, Kd, k0, n0, lane);
    const v16h b1 = load_bT_frag(BT, Kd, k0, n0 + 16, lane);
    const v16h b2 = load_bT_frag(BT, Kd, k0, n0 + 32, lane);
    const v16h b3 = load_bT_frag(BT, Kd, k0, n0 + 48, lane);
    acc0 = wmma_f16(a, b0, acc0);
    acc1 = wmma_f16(a, b1, acc1);
    acc2 = wmma_f16(a, b2, acc2);
    acc3 = wmma_f16(a, b3, acc3);
  }
  store_acc_f32(C, N, m0, n0,      lane, acc0);
  store_acc_f32(C, N, m0, n0 + 16, lane, acc1);
  store_acc_f32(C, N, m0, n0 + 32, lane, acc2);
  store_acc_f32(C, N, m0, n0 + 48, lane, acc3);
}

// ---------------------------------------------------------------------------
// Kernel 3: per-query neighborhood attention.
// Block = 256 threads (8 waves) per query.  LDS ~42 KB.
// ---------------------------------------------------------------------------
__global__ void attn_kernel(const _Float16* __restrict__ vals_h,
                            const float*    __restrict__ Q,
                            const _Float16* __restrict__ WkT,
                            const _Float16* __restrict__ WvT,
                            const int*      __restrict__ nbhd_idx,
                            const float*    __restrict__ nbhd_maskf,
                            _Float16*       __restrict__ OUT_h) {
  __shared__ __align__(32) _Float16 Nh[KS * CC];    //  8 KB gathered nbhd
  __shared__ __align__(32) _Float16 KV[KS * INNER]; // 32 KB K, reused for V
  __shared__ float simArr[HEADS * KS];              //  1 KB
  __shared__ float attnArr[HEADS * KS];             //  1 KB

  const int qi   = blockIdx.x;           // b*NN + i
  const int b    = qi >> 11;
  const int tid  = threadIdx.x;
  const int lane = tid & 31;
  const int wave = tid >> 5;

  // gather neighborhood rows (irregular hot path), 8 halfs per access
  for (int e = tid; e < KS * CC / 8; e += 256) {
    const int j   = e >> 4;                         // / (CC/8)
    const int c8  = e & 15;
    const int src = nbhd_idx[(size_t)qi * KS + j];
    *(v8h*)&Nh[j * CC + c8 * 8] =
        *(const v8h*)&vals_h[(size_t)((b << 11) + src) * CC + c8 * 8];
  }
  __syncthreads();

  // ---- K = Nh(32x128) @ Wk : 2x8 tiles of 16x64, split over 8 waves ----
  for (int t = wave; t < 16; t += 8) {
    const int m0 = (t >> 3) << 4;                   // 0 or 16
    const int n0 = (t & 7) << 6;
    v8f acc0 = {}, acc1 = {}, acc2 = {}, acc3 = {};
#pragma unroll
    for (int k0 = 0; k0 < CC; k0 += 32) {
      const v16h a  = load_a_frag(Nh, CC, m0, k0, lane);
      const v16h b0 = load_bT_frag(WkT, CC, k0, n0, lane);
      const v16h b1 = load_bT_frag(WkT, CC, k0, n0 + 16, lane);
      const v16h b2 = load_bT_frag(WkT, CC, k0, n0 + 32, lane);
      const v16h b3 = load_bT_frag(WkT, CC, k0, n0 + 48, lane);
      acc0 = wmma_f16(a, b0, acc0);
      acc1 = wmma_f16(a, b1, acc1);
      acc2 = wmma_f16(a, b2, acc2);
      acc3 = wmma_f16(a, b3, acc3);
    }
    const int n  = lane & 15;
    const int mb = m0 + (lane >> 4) * 8;
#pragma unroll
    for (int v = 0; v < 8; ++v) {
      KV[(size_t)(mb + v) * INNER + n0 + n]      = (_Float16)acc0[v];
      KV[(size_t)(mb + v) * INNER + n0 + 16 + n] = (_Float16)acc1[v];
      KV[(size_t)(mb + v) * INNER + n0 + 32 + n] = (_Float16)acc2[v];
      KV[(size_t)(mb + v) * INNER + n0 + 48 + n] = (_Float16)acc3[v];
    }
  }
  __syncthreads();

  // ---- sim + mask: one (head, j) pair per thread ----
  {
    const int h = tid >> 5;
    const int j = tid & 31;
    const float* q  = Q + (size_t)qi * INNER + h * DHEAD;
    const v8h*   kv = (const v8h*)&KV[(size_t)j * INNER + h * DHEAD];
    float s = 0.0f;
#pragma unroll
    for (int c = 0; c < DHEAD / 8; ++c) {
      const v8h kk = kv[c];
#pragma unroll
      for (int e = 0; e < 8; ++e) s += q[c * 8 + e] * (float)kk[e];
    }
    s *= 0.125f;  // DHEAD^-0.5
    const float ok = nbhd_maskf[(size_t)qi * KS + j];
    simArr[tid] = (ok > 0.5f) ? s : -3.402823466e38f;
  }
  __syncthreads();

  // ---- softmax over the 32 neighbors of each head ----
  {
    const int h = tid >> 5;
    float mx = -3.402823466e38f;
#pragma unroll
    for (int jj = 0; jj < KS; ++jj) mx = fmaxf(mx, simArr[h * KS + jj]);
    float sum = 0.0f;
#pragma unroll
    for (int jj = 0; jj < KS; ++jj) sum += __expf(simArr[h * KS + jj] - mx);
    attnArr[tid] = __expf(simArr[tid] - mx) / sum;
  }
  __syncthreads();

  // ---- V = Nh @ Wv (reuse KV buffer; prior reads fenced above) ----
  for (int t = wave; t < 16; t += 8) {
    const int m0 = (t >> 3) << 4;
    const int n0 = (t & 7) << 6;
    v8f acc0 = {}, acc1 = {}, acc2 = {}, acc3 = {};
#pragma unroll
    for (int k0 = 0; k0 < CC; k0 += 32) {
      const v16h a  = load_a_frag(Nh, CC, m0, k0, lane);
      const v16h b0 = load_bT_frag(WvT, CC, k0, n0, lane);
      const v16h b1 = load_bT_frag(WvT, CC, k0, n0 + 16, lane);
      const v16h b2 = load_bT_frag(WvT, CC, k0, n0 + 32, lane);
      const v16h b3 = load_bT_frag(WvT, CC, k0, n0 + 48, lane);
      acc0 = wmma_f16(a, b0, acc0);
      acc1 = wmma_f16(a, b1, acc1);
      acc2 = wmma_f16(a, b2, acc2);
      acc3 = wmma_f16(a, b3, acc3);
    }
    const int n  = lane & 15;
    const int mb = m0 + (lane >> 4) * 8;
#pragma unroll
    for (int v = 0; v < 8; ++v) {
      KV[(size_t)(mb + v) * INNER + n0 + n]      = (_Float16)acc0[v];
      KV[(size_t)(mb + v) * INNER + n0 + 16 + n] = (_Float16)acc1[v];
      KV[(size_t)(mb + v) * INNER + n0 + 32 + n] = (_Float16)acc2[v];
      KV[(size_t)(mb + v) * INNER + n0 + 48 + n] = (_Float16)acc3[v];
    }
  }
  __syncthreads();

  // ---- out = attn @ V : 2 consecutive outputs per thread ----
  {
    const int e0 = tid * 2;                         // 0..510
    const int h  = e0 >> 6;
    float o0 = 0.0f, o1 = 0.0f;
#pragma unroll
    for (int j = 0; j < KS; ++j) {
      const v2h vv = *(const v2h*)&KV[(size_t)j * INNER + e0];
      const float w = attnArr[h * KS + j];
      o0 += w * (float)vv[0];
      o1 += w * (float)vv[1];
    }
    v2h o; o[0] = (_Float16)o0; o[1] = (_Float16)o1;
    *(v2h*)&OUT_h[(size_t)qi * INNER + e0] = o;
  }
}

// ---------------------------------------------------------------------------
// Kernel 4: combined = OUT(8192x512) @ Wo + bo  (f16 WMMA + bias)
// ---------------------------------------------------------------------------
__global__ void gemm_bias_wmma(const _Float16* __restrict__ A,
                               const _Float16* __restrict__ BT,
                               const float* __restrict__ bias,
                               float* __restrict__ C,
                               int M, int Kd, int N) {
  const int lane    = threadIdx.x & 31;
  const int wave    = threadIdx.x >> 5;
  const int tiles_n = N >> 6;
  const int tile    = blockIdx.x * (blockDim.x >> 5) + wave;
  if (tile >= (M >> 4) * tiles_n) return;
  const int m0 = (tile / tiles_n) << 4;
  const int n0 = (tile % tiles_n) << 6;

  v8f acc0 = {}, acc1 = {}, acc2 = {}, acc3 = {};
  for (int k0 = 0; k0 < Kd; k0 += 32) {
    const v16h a  = load_a_frag(A, Kd, m0, k0, lane);
    const v16h b0 = load_bT_frag(BT, Kd, k0, n0, lane);
    const v16h b1 = load_bT_frag(BT, Kd, k0, n0 + 16, lane);
    const v16h b2 = load_bT_frag(BT, Kd, k0, n0 + 32, lane);
    const v16h b3 = load_bT_frag(BT, Kd, k0, n0 + 48, lane);
    acc0 = wmma_f16(a, b0, acc0);
    acc1 = wmma_f16(a, b1, acc1);
    acc2 = wmma_f16(a, b2, acc2);
    acc3 = wmma_f16(a, b3, acc3);
  }
  const int n  = lane & 15;
  const int mb = m0 + (lane >> 4) * 8;
  const float bn0 = bias[n0 + n];
  const float bn1 = bias[n0 + 16 + n];
  const float bn2 = bias[n0 + 32 + n];
  const float bn3 = bias[n0 + 48 + n];
#pragma unroll
  for (int v = 0; v < 8; ++v) {
    C[(size_t)(mb + v) * N + n0 + n]      = acc0[v] + bn0;
    C[(size_t)(mb + v) * N + n0 + 16 + n] = acc1[v] + bn1;
    C[(size_t)(mb + v) * N + n0 + 32 + n] = acc2[v] + bn2;
    C[(size_t)(mb + v) * N + n0 + 48 + n] = acc3[v] + bn3;
  }
}

// ---------------------------------------------------------------------------
// Host-side launcher
// ---------------------------------------------------------------------------
extern "C" void kernel_launch(void* const* d_in, const int* in_sizes, int n_in,
                              void* d_out, int out_size, void* d_ws, size_t ws_size,
                              hipStream_t stream) {
  (void)in_sizes; (void)n_in; (void)out_size; (void)ws_size;

  const float*         pairs = (const float*)d_in[0];
  const float*         vals  = (const float*)d_in[1];
  const unsigned char* mask  = (const unsigned char*)d_in[2];
  const float*         noise = (const float*)d_in[3];
  const float*         Wq    = (const float*)d_in[4];
  const float*         Wk    = (const float*)d_in[5];
  const float*         Wv    = (const float*)d_in[6];
  const float*         Wo    = (const float*)d_in[7];
  const float*         bo    = (const float*)d_in[8];

  // output layout: pairs (pass-through) | combined | mask
  const size_t pairs_elems    = (size_t)BB * NN * NN * DD;   // 100,663,296
  const size_t combined_elems = (size_t)BB * NN * CC;        //   1,048,576
  float* out_pairs    = (float*)d_out;
  float* out_combined = out_pairs + pairs_elems;
  float* out_mask     = out_combined + combined_elems;

  // workspace layout (bytes)
  char* ws = (char*)d_ws;
  size_t off = 0;
  int*      nbhd_idx   = (int*)(ws + off);       off += (size_t)BB * NN * KS * 4;
  float*    nbhd_maskf = (float*)(ws + off);     off += (size_t)BB * NN * KS * 4;
  _Float16* vals_h     = (_Float16*)(ws + off);  off += (size_t)BB * NN * CC * 2;
  _Float16* WqT        = (_Float16*)(ws + off);  off += (size_t)CC * INNER * 2;
  _Float16* WkT        = (_Float16*)(ws + off);  off += (size_t)CC * INNER * 2;
  _Float16* WvT        = (_Float16*)(ws + off);  off += (size_t)CC * INNER * 2;
  _Float16* WoT        = (_Float16*)(ws + off);  off += (size_t)INNER * CC * 2;
  float*    Qbuf       = (float*)(ws + off);     off += (size_t)BB * NN * INNER * 4;
  _Float16* OUT_h      = (_Float16*)(ws + off);  off += (size_t)BB * NN * INNER * 2;

  const int nq = BB * NN;  // 8192 query rows

  // 0) precision conversion (weights transposed) + mask tail
  prep_kernel<<<4096, 256, 0, stream>>>(vals, Wq, Wk, Wv, Wo, mask,
                                        vals_h, WqT, WkT, WvT, WoT, out_mask);

  // 1) scores + top-32 ball query, fused pairs pass-through copy
  topk_kernel<<<nq, 256, 0, stream>>>(pairs, mask, noise,
                                      out_pairs, nbhd_idx, nbhd_maskf);

  // 2) Q = vals @ Wq   (8192x128 @ 128x512): 4096 wave-tiles of 16x64
  gemm_f16_wmma<<<512, 256, 0, stream>>>(vals_h, WqT, Qbuf, nq, CC, INNER);

  // 3) gather + K/V projection + local attention, one block per query
  attn_kernel<<<nq, 256, 0, stream>>>(vals_h, Qbuf, WkT, WvT,
                                      nbhd_idx, nbhd_maskf, OUT_h);

  // 4) combined = OUT @ Wo + bo  (8192x512 @ 512x128): 1024 wave-tiles
  gemm_bias_wmma<<<128, 256, 0, stream>>>(OUT_h, WoT, bo, out_combined,
                                          nq, INNER, CC);
}
